// Qwen3HybridAttentionDecoderLayer_48996986913107
// MI455X (gfx1250) — matrix operand order repaired and checked by reading, not compiled, not run on hardware
//
#include <hip/hip_runtime.h>
#include <hip/hip_bf16.h>

// ---- problem constants (match reference) ----
#define T_    2048
#define H_    2048
#define NH_   32
#define NKV_  4
#define HD_   128
#define QKVN_ 5120           // (NH + 2*NKV) * HD
#define E_    16
#define IM_   1024
#define TOPK_ 2
#define SLOTS_ (T_ * TOPK_)

typedef __attribute__((ext_vector_type(16))) __bf16 bf16x16;
typedef __attribute__((ext_vector_type(8)))  __bf16 bf16x8;
typedef __attribute__((ext_vector_type(8)))  float  v8f;
typedef __attribute__((ext_vector_type(4)))  unsigned uint32x4;
typedef __attribute__((ext_vector_type(8)))  int      int32x8;
typedef __attribute__((ext_vector_type(4)))  int      int32x4;

#if defined(__HIP_DEVICE_COMPILE__) && __has_builtin(__builtin_amdgcn_tensor_load_to_lds)
#define HAVE_TDM 1
#else
#define HAVE_TDM 0
#endif

__device__ inline __bf16 f2bf(float f) { return (__bf16)f; }
__device__ inline float  bf2f(__bf16 b) { return (float)b; }

// pack two floats into two bf16 in one dword (lets compiler use v_cvt_pk_bf16_f32)
__device__ inline unsigned pack2bf(float a, float b) {
  __bf16 ba = (__bf16)a, bb = (__bf16)b;
  unsigned short ua, ub;
  __builtin_memcpy(&ua, &ba, 2);
  __builtin_memcpy(&ub, &bb, 2);
  return (unsigned)ua | ((unsigned)ub << 16);
}

__device__ inline v8f vzero8() {
  v8f z;
#pragma unroll
  for (int i = 0; i < 8; i++) z[i] = 0.f;
  return z;
}

// Build a 16x32 (16-bit) A-fragment for one lane from two contiguous 8-elem
// chunks (contiguous k order: lanes 0-15 hold K{0..15},
// lanes 16-31 hold K{16..31}; caller passes lane-adjusted pointers).
__device__ inline bf16x16 pack16(const __bf16* p0, const __bf16* p1) {
  bf16x8 a = *(const bf16x8*)p0;
  bf16x8 b = *(const bf16x8*)p1;
  bf16x16 r;
#pragma unroll
  for (int i = 0; i < 8; i++) { r[i] = a[i]; r[i + 8] = b[i]; }
  return r;
}

__device__ inline v8f wmma_bf(bf16x16 a, bf16x16 b, v8f c) {
  // (neg_a, A, neg_b, B, c_mod, C, reuse_a, reuse_b)
  return __builtin_amdgcn_wmma_f32_16x16x32_bf16(false, a, false, b, (short)0, c,
                                                 false, false);
}

// CDNA5 async global->LDS copy (16B per lane), tracked by ASYNCcnt.
typedef __attribute__((address_space(3))) __bf16 lds_bf16_t;
__device__ inline void async_copy_b128(__bf16* lds_dst, const __bf16* gsrc) {
  lds_bf16_t* lp = (lds_bf16_t*)lds_dst;
  asm volatile("global_load_async_to_lds_b128 %0, %1, off"
               :: "v"(lp), "v"(gsrc)
               : "memory");
}
__device__ inline void async_wait0() {
  asm volatile("s_wait_asynccnt 0" ::: "memory");
}

__device__ inline unsigned lds_addr_of(const __bf16* p) {
  return (unsigned)(unsigned long long)(lds_bf16_t*)p;
}

__device__ inline void tensor_wait0() {
#if defined(__HIP_DEVICE_COMPILE__) && __has_builtin(__builtin_amdgcn_s_wait_tensorcnt)
  __builtin_amdgcn_s_wait_tensorcnt(0);
#else
  asm volatile("s_wait_tensorcnt 0" ::: "memory");
#endif
}

#if HAVE_TDM
// Tensor Data Mover: DMA one 2D tile (tile_w x tile_h, 16-bit elements,
// row stride in elements) from global memory into LDS. D# packed per
// CDNA5 ISA (group0: count/lds/global/type=2,
// group1: data_size=2B + tensor/tile dims + dim0 stride; groups 2/3 = 0).
__device__ inline void tdm_load_tile_2d(unsigned lds_addr, const void* gaddr,
                                        unsigned tile_w, unsigned tile_h,
                                        long long row_stride_elems) {
  unsigned long long ga = (unsigned long long)gaddr;
  uint32x4 g0;
  g0[0] = 1u;                                         // count=1 (user, load)
  g0[1] = lds_addr;                                   // lds byte address
  g0[2] = (unsigned)(ga & 0xFFFFFFFFull);             // global addr lo
  g0[3] = (unsigned)((ga >> 32) & 0x01FFFFFFull) | (2u << 30);  // hi + type=2
  unsigned long long st = (unsigned long long)row_stride_elems;
  int32x8 g1;
  g1[0] = (int)(1u << 16);                            // data_size = 2 bytes
  g1[1] = (int)((tile_w & 0xFFFFu) << 16);            // tensor_dim0 lo16 @63:48
  g1[2] = (int)((tile_w >> 16) | ((tile_h & 0xFFFFu) << 16));  // dim0 hi / dim1 lo
  g1[3] = (int)((tile_h >> 16) | (tile_w << 16));     // dim1 hi / tile_dim0
  g1[4] = (int)(tile_h & 0xFFFFu);                    // tile_dim1 (tile_dim2=0)
  g1[5] = (int)(unsigned)(st & 0xFFFFFFFFull);        // dim0 stride lo32
  g1[6] = (int)(unsigned)((st >> 32) & 0xFFFFull);    // dim0 stride hi16
  g1[7] = 0;
  int32x4 z4 = {0, 0, 0, 0};
#if __clang_major__ >= 23
  int32x8 z8 = {0, 0, 0, 0, 0, 0, 0, 0};
  __builtin_amdgcn_tensor_load_to_lds(g0, g1, z4, z4, z8, 0);
#else
  __builtin_amdgcn_tensor_load_to_lds(g0, g1, z4, z4, 0);
#endif
}
#endif

// =====================================================================
// zero / prefix helpers
// =====================================================================
__global__ void zero_k(float* __restrict__ p, int n, int* __restrict__ c, int ec) {
  int i = blockIdx.x * blockDim.x + threadIdx.x;
  if (i < n) p[i] = 0.f;
  if (i < ec) c[i] = 0;
}

__global__ void prefix_k(const int* __restrict__ cnt, int* __restrict__ off) {
  if (threadIdx.x == 0 && blockIdx.x == 0) {
    int s = 0;
    for (int e = 0; e < E_; e++) { off[e] = s; s += cnt[e]; }
  }
}

// =====================================================================
// (a [+ b]) -> res_out (f32, optional), RMSNorm*w -> bf16
// =====================================================================
__global__ __launch_bounds__(256) void add_rms_k(
    const float* __restrict__ a, const float* __restrict__ b,
    const float* __restrict__ w, float* __restrict__ sum_out,
    __bf16* __restrict__ nrm_out) {
  int t = blockIdx.x, tid = threadIdx.x;
  const float* ar = a + (size_t)t * H_;
  const float* br = b ? (b + (size_t)t * H_) : nullptr;
  float v[8];
  float ss = 0.f;
#pragma unroll
  for (int i = 0; i < 8; i++) {
    int j = tid + i * 256;
    float x = ar[j];
    if (br) x += br[j];
    v[i] = x;
    ss += x * x;
    if (sum_out) sum_out[(size_t)t * H_ + j] = x;
  }
#pragma unroll
  for (int o = 1; o < 32; o <<= 1) ss += __shfl_xor(ss, o, 32);
  __shared__ float sm[8];
  if ((tid & 31) == 0) sm[tid >> 5] = ss;
  __syncthreads();
  float tot = 0.f;
#pragma unroll
  for (int i = 0; i < 8; i++) tot += sm[i];
  float rs = rsqrtf(tot / (float)H_ + 1e-6f);
#pragma unroll
  for (int i = 0; i < 8; i++) {
    int j = tid + i * 256;
    nrm_out[(size_t)t * H_ + j] = f2bf(v[i] * rs * w[j]);
  }
}

// =====================================================================
// qkv post: per-head RMSNorm (q/k) + RoPE, v passthrough, all -> bf16
// grid (T, NH + 2*NKV), block 128
// =====================================================================
__global__ __launch_bounds__(128) void norm_rope_k(
    const float* __restrict__ qkv, const int* __restrict__ positions,
    const float* __restrict__ qw, const float* __restrict__ kw,
    __bf16* __restrict__ qo, __bf16* __restrict__ ko, __bf16* __restrict__ vo) {
  int t = blockIdx.x, hh = blockIdx.y, d = threadIdx.x;
  const float* row = qkv + (size_t)t * QKVN_;
  if (hh >= NH_ + NKV_) {  // v heads: just convert
    int vh = hh - (NH_ + NKV_);
    vo[(size_t)t * (NKV_ * HD_) + vh * HD_ + d] =
        f2bf(row[(NH_ + NKV_) * HD_ + vh * HD_ + d]);
    return;
  }
  bool isq = hh < NH_;
  int off = isq ? hh * HD_ : NH_ * HD_ + (hh - NH_) * HD_;
  const float* w = isq ? qw : kw;
  float x = row[off + d];
  float ss = x * x;
#pragma unroll
  for (int o = 1; o < 32; o <<= 1) ss += __shfl_xor(ss, o, 32);
  __shared__ float sm[4];
  __shared__ float xn[HD_];
  if ((d & 31) == 0) sm[d >> 5] = ss;
  __syncthreads();
  float tot = sm[0] + sm[1] + sm[2] + sm[3];
  float rs = rsqrtf(tot / (float)HD_ + 1e-6f);
  xn[d] = x * rs * w[d];
  __syncthreads();
  if (d < 64) {
    float pos = (float)positions[t];
    float inv = __powf(10000.f, -(float)d / 64.f);
    float fr = pos * inv;
    float s, c;
    __sincosf(fr, &s, &c);
    float x1 = xn[d], x2 = xn[d + 64];
    __bf16* dst = isq ? (qo + (size_t)t * (NH_ * HD_) + hh * HD_)
                      : (ko + (size_t)t * (NKV_ * HD_) + (hh - NH_) * HD_);
    dst[d]      = f2bf(x1 * c - x2 * s);
    dst[d + 64] = f2bf(x2 * c + x1 * s);
  }
}

// =====================================================================
// Unified bf16 WMMA GEMM: C[M,N] += A[M,K](bf16) * B[K,N](fp32, cvt->bf16)
// Block tile 128x128x32, 256 threads (8 waves), wave tile 64x32.
// Double-buffered LDS. Dense (non-gathered) A tiles are DMA'd by the
// Tensor Data Mover (TENSORcnt); B tiles are fp32->bf16 converted and
// transposed through registers; weight stream prefetched ahead.
// EPI: 0 = store f32; 1 = +addend, store f32; 2 = store bf16 (compact rows);
//      3 = silu(acc)*up_buf -> bf16 (compact rows); 4 = atomic weighted scatter.
// Expert mode (cntp != null): blockIdx.z = expert, M_eff = cnt[e],
// compact C rows at off[e], B advanced by e*bstride, gather/sw/stok by e*gstride.
// =====================================================================
template <int EPI>
__global__ __launch_bounds__(256) void gemm_k(
    const __bf16* __restrict__ A, int lda, const float* __restrict__ Bmat, int ldb,
    int M, int K,
    const int* __restrict__ gather, int gstride,
    const int* __restrict__ cntp, const int* __restrict__ offp, long long bstride,
    float* __restrict__ Cf, int ldc, const float* __restrict__ addf,
    __bf16* __restrict__ Cb, int ldcb, const __bf16* __restrict__ upbuf,
    float* __restrict__ scat, const float* __restrict__ swt,
    const int* __restrict__ stok) {
  constexpr bool TDM = (HAVE_TDM != 0) && (EPI <= 1);  // dense A tiles only
  __shared__ __align__(32) __bf16 sA[2][128 * 32];
  __shared__ __align__(32) __bf16 sB[2][128 * 32];  // transposed: [n][k]

  int tid = threadIdx.x;
  int e = blockIdx.z;
  int Me = M, cbase = 0;
  const float* Bp = Bmat;
  const int* g = gather;
  const float* swp = swt;
  const int* stp = stok;
  if (cntp) {
    Me = cntp[e];
    cbase = offp[e];
    Bp = Bmat + (long long)e * bstride;
    if (g) g += (long long)e * gstride;
    if (swt) { swp = swt + (long long)e * gstride; stp = stok + (long long)e * gstride; }
  }
  int mb = blockIdx.y * 128;
  if (mb >= Me) return;
  int nb = blockIdx.x * 128;
  int lane = tid & 31, wave = tid >> 5;
  int wm = wave >> 2, wn = wave & 3;
  int lm = lane & 15, hi = (lane >> 4) & 1;

  v8f acc[4][2];
#pragma unroll
  for (int ms = 0; ms < 4; ms++)
#pragma unroll
    for (int nt = 0; nt < 2; nt++) acc[ms][nt] = vzero8();

  // A staging (manual path): 2 threads per row, 16 bf16 (32B) each
  int ar_loc = mb + (tid >> 1);
  int ha = (tid & 1) * 16;
  bool a_ok = (ar_loc < Me);
  int a_src;
  if (g)         a_src = a_ok ? g[ar_loc] : 0;
  else if (cntp) a_src = cbase + (a_ok ? ar_loc : 0);
  else           a_src = a_ok ? ar_loc : 0;
  const __bf16* aptr = A + (size_t)a_src * lda + ha;
  // A staging (TDM path): whole 128x32 tile descriptor, issued by wave 0
  const __bf16* atile = A + (size_t)mb * lda;
  unsigned ldsA[2];
  ldsA[0] = lds_addr_of(&sA[0][0]);
  ldsA[1] = lds_addr_of(&sA[1][0]);

  // B staging: thread owns a K-pair x 8 columns; packed bf16x2 dword LDS stores
  int kp = tid >> 4;            // 0..15 -> k = {2kp, 2kp+1}
  int n0 = (tid & 15) * 8;      // 8 columns
  const float* bptr0 = Bp + (size_t)(2 * kp) * ldb + nb + n0;
  const float* bptr1 = bptr0 + ldb;

  uint4 ra0, ra1;
  float4 rb00, rb01, rb10, rb11;

  auto ldg = [&](int k0) {
    if (!TDM) {
      uint4 z = {0, 0, 0, 0};
      ra0 = z; ra1 = z;
      if (a_ok) {
        const uint4* s = (const uint4*)(aptr + k0);
        ra0 = s[0];
        ra1 = s[1];
      }
    }
    const float4* s0 = (const float4*)(bptr0 + (size_t)k0 * ldb);
    const float4* s1 = (const float4*)(bptr1 + (size_t)k0 * ldb);
    rb00 = s0[0]; rb01 = s0[1];
    rb10 = s1[0]; rb11 = s1[1];
  };
  auto stlds = [&](int buf) {
    if (!TDM) {
      uint4* da = (uint4*)&sA[buf][(tid >> 1) * 32 + ha];
      da[0] = ra0;
      da[1] = ra1;
    }
    float r0[8] = {rb00.x, rb00.y, rb00.z, rb00.w, rb01.x, rb01.y, rb01.z, rb01.w};
    float r1[8] = {rb10.x, rb10.y, rb10.z, rb10.w, rb11.x, rb11.y, rb11.z, rb11.w};
    unsigned* db = (unsigned*)&sB[buf][0];
#pragma unroll
    for (int i = 0; i < 8; i++)
      db[(n0 + i) * 16 + kp] = pack2bf(r0[i], r1[i]);
  };

#if HAVE_TDM
  if (TDM && wave == 0) tdm_load_tile_2d(ldsA[0], atile, 32, 128, lda);
#endif
  ldg(0);
  stlds(0);
  if (TDM && wave == 0) tensor_wait0();
  __syncthreads();
  int cur = 0;
  for (int k0 = 0; k0 < K; k0 += 32) {
    bool more = (k0 + 32) < K;
    if (more) {
#if HAVE_TDM
      if (TDM && wave == 0)
        tdm_load_tile_2d(ldsA[cur ^ 1], atile + (k0 + 32), 32, 128, lda);
#endif
      ldg(k0 + 32);
      if (k0 + 64 < K) __builtin_prefetch(bptr0 + (size_t)(k0 + 64) * ldb, 0, 0);
    }
    bf16x16 bfr[2];
#pragma unroll
    for (int nt = 0; nt < 2; nt++) {
      int n = wn * 32 + nt * 16 + lm;
      bfr[nt] = *(const bf16x16*)&sB[cur][n * 32 + hi * 16];
    }
#pragma unroll
    for (int ms = 0; ms < 4; ms++) {
      const __bf16* rp = &sA[cur][(wm * 64 + ms * 16 + lm) * 32 + hi * 16];
      bf16x16 af = pack16(rp, rp + 8);
#pragma unroll
      for (int nt = 0; nt < 2; nt++) acc[ms][nt] = wmma_bf(af, bfr[nt], acc[ms][nt]);
    }
    if (more) {
      stlds(cur ^ 1);
      if (TDM && wave == 0) tensor_wait0();
      __syncthreads();
      cur ^= 1;
    }
  }

  // epilogue (C layout: VGPR j -> row j (lanes 0-15) / j+8 (lanes 16-31))
#pragma unroll
  for (int ms = 0; ms < 4; ms++)
#pragma unroll
    for (int nt = 0; nt < 2; nt++)
#pragma unroll
      for (int j = 0; j < 8; j++) {
        int rl = mb + wm * 64 + ms * 16 + j + hi * 8;
        if (rl >= Me) continue;
        int col = nb + wn * 32 + nt * 16 + lm;
        float v = acc[ms][nt][j];
        if (EPI == 0) {
          Cf[(size_t)rl * ldc + col] = v;
        } else if (EPI == 1) {
          Cf[(size_t)rl * ldc + col] = v + addf[(size_t)rl * ldc + col];
        } else if (EPI == 2) {
          Cb[(size_t)(cbase + rl) * ldcb + col] = f2bf(v);
        } else if (EPI == 3) {
          size_t o = (size_t)(cbase + rl) * ldcb + col;
          float u = bf2f(upbuf[o]);
          float gs = v / (1.f + __expf(-v));
          Cb[o] = f2bf(gs * u);
        } else {  // EPI == 4
          int tok = stp[rl];
          float w = swp[rl];
          atomicAdd(&scat[(size_t)tok * ldc + col], w * v);
        }
      }
}

// =====================================================================
// Flash attention, WMMA bf16. grid (NH, T/64), block 128 (4 waves x 16 q rows).
// K tile staged with CDNA5 async global->LDS copies (ASYNCcnt).
// =====================================================================
__global__ __launch_bounds__(128) void attn_k(
    const __bf16* __restrict__ Q, const __bf16* __restrict__ Kc,
    const __bf16* __restrict__ Vc, __bf16* __restrict__ O) {
  int h = blockIdx.x;
  int qb = blockIdx.y * 64;
  int kvh = h >> 3;  // NH/NKV = 8
  int tid = threadIdx.x, lane = tid & 31, wave = tid >> 5;
  int lm = lane & 15, hi = (lane >> 4) & 1;

  __shared__ __align__(32) __bf16 sK[32 * HD_];   // [key][hd]
  __shared__ __align__(32) __bf16 sVt[HD_ * 32];  // [hd][key]
  __shared__ __align__(32) __bf16 sP[4][16 * 32]; // per-wave P tile

  int qw = qb + wave * 16;
  int qrow = qw + lm;
  const __bf16* qp = Q + (size_t)qrow * (NH_ * HD_) + h * HD_;
  bf16x16 qf[4];
#pragma unroll
  for (int kc = 0; kc < 4; kc++) {
    const __bf16* p = qp + kc * 32 + hi * 16;
    qf[kc] = pack16(p, p + 8);
  }
  v8f o[8];
#pragma unroll
  for (int i = 0; i < 8; i++) o[i] = vzero8();
  float mrow[8], lrow[8];
#pragma unroll
  for (int j = 0; j < 8; j++) { mrow[j] = -1e30f; lrow[j] = 0.f; }
  const float scale = 0.08838834764831845f;  // 1/sqrt(128)

  for (int kt = 0; kt < qb + 64; kt += 32) {
    {  // stage K tile (32x128 bf16) via async copies: 16B per lane x 4
      int r = tid >> 2;
      int c = (tid & 3) * 32;
      const __bf16* gsrc = Kc + (size_t)(kt + r) * (NKV_ * HD_) + kvh * HD_ + c;
      __bf16* ldst = &sK[r * HD_ + c];
#pragma unroll
      for (int i = 0; i < 4; i++) async_copy_b128(ldst + 8 * i, gsrc + 8 * i);
    }
    {  // stage V transposed: key-pairs packed into dword LDS stores
      int p = tid & 15;           // key pair -> keys {2p, 2p+1}
      int hb = (tid >> 4) * 16;   // hd block of 16
      const __bf16* s0 = Vc + (size_t)(kt + 2 * p) * (NKV_ * HD_) + kvh * HD_ + hb;
      const __bf16* s1 = s0 + (NKV_ * HD_);
      unsigned short u0[16], u1[16];
      *(uint4*)&u0[0] = *(const uint4*)s0;
      *(uint4*)&u0[8] = *(const uint4*)(s0 + 8);
      *(uint4*)&u1[0] = *(const uint4*)s1;
      *(uint4*)&u1[8] = *(const uint4*)(s1 + 8);
      unsigned* dst = (unsigned*)&sVt[0];
#pragma unroll
      for (int i = 0; i < 16; i++)
        dst[(hb + i) * 16 + p] = (unsigned)u0[i] | ((unsigned)u1[i] << 16);
    }
    async_wait0();
    __syncthreads();
    if (kt <= qw + 15) {  // tile has at least one causal key for this wave
      v8f s0 = vzero8(), s1 = vzero8();
#pragma unroll
      for (int kc = 0; kc < 4; kc++) {
        bf16x16 b0 = *(const bf16x16*)&sK[(lm) * HD_ + kc * 32 + hi * 16];
        bf16x16 b1 = *(const bf16x16*)&sK[(16 + lm) * HD_ + kc * 32 + hi * 16];
        s0 = wmma_bf(qf[kc], b0, s0);
        s1 = wmma_bf(qf[kc], b1, s1);
      }
#pragma unroll
      for (int j = 0; j < 8; j++) {
        int qr = qw + j + hi * 8;
        float a0 = s0[j] * scale, a1 = s1[j] * scale;
        if (kt + lm > qr) a0 = -1e30f;
        if (kt + 16 + lm > qr) a1 = -1e30f;
        float rmax = fmaxf(a0, a1);
#pragma unroll
        for (int os = 1; os < 16; os <<= 1) rmax = fmaxf(rmax, __shfl_xor(rmax, os, 32));
        float mn = fmaxf(mrow[j], rmax);
        float alpha = __expf(mrow[j] - mn);
        mrow[j] = mn;
        float p0 = __expf(a0 - mn), p1 = __expf(a1 - mn);
        float rs = p0 + p1;
#pragma unroll
        for (int os = 1; os < 16; os <<= 1) rs += __shfl_xor(rs, os, 32);
        lrow[j] = lrow[j] * alpha + rs;
#pragma unroll
        for (int nt = 0; nt < 8; nt++) o[nt][j] *= alpha;
        int prow = j + hi * 8;
        sP[wave][prow * 32 + lm] = f2bf(p0);
        sP[wave][prow * 32 + 16 + lm] = f2bf(p1);
      }
      const __bf16* pp = &sP[wave][lm * 32 + hi * 16];
      bf16x16 pf = pack16(pp, pp + 8);
#pragma unroll
      for (int nt = 0; nt < 8; nt++) {
        bf16x16 bv = *(const bf16x16*)&sVt[(nt * 16 + lm) * 32 + hi * 16];
        o[nt] = wmma_bf(pf, bv, o[nt]);
      }
    }
    __syncthreads();
  }
#pragma unroll
  for (int j = 0; j < 8; j++) {
    float inv = 1.f / lrow[j];
    int row = qw + j + hi * 8;
    __bf16* dst = O + (size_t)row * (NH_ * HD_) + h * HD_;
#pragma unroll
    for (int nt = 0; nt < 8; nt++) dst[nt * 16 + lm] = f2bf(o[nt][j] * inv);
  }
}

// =====================================================================
// Router: logits = h2 @ router_w, softmax, top-2, normalized weights,
// build per-expert token lists via atomics. grid T, block 256.
// =====================================================================
__global__ __launch_bounds__(256) void router_k(
    const __bf16* __restrict__ h2, const float* __restrict__ rw,
    int* __restrict__ cnt, int* __restrict__ etok, float* __restrict__ ew) {
  int t = blockIdx.x, tid = threadIdx.x;
  float acc[E_];
#pragma unroll
  for (int e = 0; e < E_; e++) acc[e] = 0.f;
  for (int j = tid; j < H_; j += 256) {
    float x = bf2f(h2[(size_t)t * H_ + j]);
    const float* r = rw + (size_t)j * E_;
#pragma unroll
    for (int e = 0; e < E_; e++) acc[e] += x * r[e];
  }
  __shared__ float red[256];
  __shared__ float logit[E_];
  for (int e = 0; e < E_; e++) {
    red[tid] = acc[e];
    __syncthreads();
    for (int s = 128; s > 0; s >>= 1) {
      if (tid < s) red[tid] += red[tid + s];
      __syncthreads();
    }
    if (tid == 0) logit[e] = red[0];
    __syncthreads();
  }
  if (tid == 0) {
    float m = -1e30f;
    for (int e = 0; e < E_; e++) m = fmaxf(m, logit[e]);
    float p[E_];
    for (int e = 0; e < E_; e++) p[e] = __expf(logit[e] - m);
    int i0 = 0;
    for (int e = 1; e < E_; e++) if (p[e] > p[i0]) i0 = e;
    int i1 = (i0 == 0) ? 1 : 0;
    for (int e = 0; e < E_; e++) if (e != i0 && p[e] > p[i1]) i1 = e;
    float s2 = p[i0] + p[i1];
    int s0 = atomicAdd(&cnt[i0], 1);
    etok[i0 * T_ + s0] = t; ew[i0 * T_ + s0] = p[i0] / s2;
    int s1 = atomicAdd(&cnt[i1], 1);
    etok[i1 * T_ + s1] = t; ew[i1 * T_ + s1] = p[i1] / s2;
  }
}

// =====================================================================
extern "C" void kernel_launch(void* const* d_in, const int* in_sizes, int n_in,
                              void* d_out, int out_size, void* d_ws, size_t ws_size,
                              hipStream_t stream) {
  const int*   positions = (const int*)d_in[0];
  const float* hs    = (const float*)d_in[1];
  const float* resid = (const float*)d_in[2];
  const float* ln1   = (const float*)d_in[3];
  const float* ln2   = (const float*)d_in[4];
  const float* w_qkv = (const float*)d_in[5];
  const float* w_o   = (const float*)d_in[6];
  const float* qn    = (const float*)d_in[7];
  const float* kn    = (const float*)d_in[8];
  const float* rw    = (const float*)d_in[9];
  const float* wg    = (const float*)d_in[10];
  const float* wu    = (const float*)d_in[11];
  const float* wd    = (const float*)d_in[12];
  (void)in_sizes; (void)n_in; (void)out_size; (void)ws_size;

  float* mlp_out = (float*)d_out;                       // [T,H]
  float* res2    = (float*)d_out + (size_t)T_ * H_;     // [T,H]

  char* ws = (char*)d_ws;
  auto alloc = [&](size_t bytes) {
    char* p = ws;
    ws += (bytes + 255) & ~(size_t)255;
    return p;
  };
  float*   res1  = (float*)alloc((size_t)T_ * H_ * 4);
  __bf16*  h1    = (__bf16*)alloc((size_t)T_ * H_ * 2);
  float*   qkv   = (float*)alloc((size_t)T_ * QKVN_ * 4);
  __bf16*  qb    = (__bf16*)alloc((size_t)T_ * NH_ * HD_ * 2);
  __bf16*  kb    = (__bf16*)alloc((size_t)T_ * NKV_ * HD_ * 2);
  __bf16*  vb    = (__bf16*)alloc((size_t)T_ * NKV_ * HD_ * 2);
  __bf16*  attnb = (__bf16*)alloc((size_t)T_ * NH_ * HD_ * 2);
  __bf16*  h2    = (__bf16*)alloc((size_t)T_ * H_ * 2);
  int*     cnt   = (int*)alloc(E_ * 4);
  int*     off   = (int*)alloc(E_ * 4);
  int*     etok  = (int*)alloc((size_t)E_ * T_ * 4);
  float*   ew    = (float*)alloc((size_t)E_ * T_ * 4);
  __bf16*  upb   = (__bf16*)alloc((size_t)SLOTS_ * IM_ * 2);
  __bf16*  prodb = (__bf16*)alloc((size_t)SLOTS_ * IM_ * 2);

  dim3 b256(256), b128(128);

  // 0) zero mlp_out + expert counts
  zero_k<<<dim3((T_ * H_ + 255) / 256), b256, 0, stream>>>(mlp_out, T_ * H_, cnt, E_);
  // 1) res1 = hs + residual; h1 = RMS(res1)*ln1 (bf16)
  add_rms_k<<<dim3(T_), b256, 0, stream>>>(hs, resid, ln1, res1, h1);
  // 2) qkv = h1 @ w_qkv   (A tile via TDM)
  gemm_k<0><<<dim3(QKVN_ / 128, T_ / 128, 1), b256, 0, stream>>>(
      h1, H_, w_qkv, QKVN_, T_, H_, nullptr, 0, nullptr, nullptr, 0,
      qkv, QKVN_, nullptr, nullptr, 0, nullptr, nullptr, nullptr, nullptr);
  // 3) per-head RMS + RoPE -> q/k/v bf16
  norm_rope_k<<<dim3(T_, NH_ + 2 * NKV_), b128, 0, stream>>>(qkv, positions, qn, kn,
                                                             qb, kb, vb);
  // 4) causal GQA flash attention
  attn_k<<<dim3(NH_, T_ / 64), b128, 0, stream>>>(qb, kb, vb, attnb);
  // 5) res2 = attn @ w_o + res1  (written straight into d_out; A tile via TDM)
  gemm_k<1><<<dim3(H_ / 128, T_ / 128, 1), b256, 0, stream>>>(
      attnb, NH_ * HD_, w_o, H_, T_, NH_ * HD_, nullptr, 0, nullptr, nullptr, 0,
      res2, H_, res1, nullptr, 0, nullptr, nullptr, nullptr, nullptr);
  // 6) h2 = RMS(res2)*ln2 (bf16)
  add_rms_k<<<dim3(T_), b256, 0, stream>>>(res2, nullptr, ln2, nullptr, h2);
  // 7) router + top-2 -> per-expert token lists
  router_k<<<dim3(T_), b256, 0, stream>>>(h2, rw, cnt, etok, ew);
  // 8) exclusive prefix over counts -> compact slot offsets
  prefix_k<<<dim3(1), dim3(32), 0, stream>>>(cnt, off);
  // 9) up[e] = gather(h2) @ w_up[e]    (bf16, compact rows)
  gemm_k<2><<<dim3(IM_ / 128, T_ / 128, E_), b256, 0, stream>>>(
      h2, H_, wu, IM_, T_, H_, etok, T_, cnt, off, (long long)H_ * IM_,
      nullptr, 0, nullptr, upb, IM_, nullptr, nullptr, nullptr, nullptr);
  // 10) prod[e] = silu(gather(h2) @ w_gate[e]) * up[e]
  gemm_k<3><<<dim3(IM_ / 128, T_ / 128, E_), b256, 0, stream>>>(
      h2, H_, wg, IM_, T_, H_, etok, T_, cnt, off, (long long)H_ * IM_,
      nullptr, 0, nullptr, prodb, IM_, upb, nullptr, nullptr, nullptr);
  // 11) mlp_out[token] += w * (prod[e] @ w_down[e])   (atomic scatter)
  gemm_k<4><<<dim3(H_ / 128, T_ / 128, E_), b256, 0, stream>>>(
      prodb, IM_, wd, H_, T_, IM_, nullptr, T_, cnt, off, (long long)IM_ * H_,
      nullptr, H_, nullptr, nullptr, 0, nullptr, mlp_out, ew, etok);
}
